// TTLayerConv_20134806683693
// MI455X (gfx1250) — matrix operand order repaired and checked by reading, not compiled, not run on hardware
//
#include <hip/hip_runtime.h>

typedef __attribute__((ext_vector_type(16))) _Float16 v16h;
typedef __attribute__((ext_vector_type(8)))  _Float16 v8h;
typedef __attribute__((ext_vector_type(8)))  float    v8f;

#define LDSTR 1032  // 1024 channels + 8-half pad (keeps 16B align, spreads LDS banks)

// Async 16B global -> LDS copy (CDNA5, tracked by ASYNCcnt, no VGPR round-trip)
__device__ __forceinline__ void async_cp16(unsigned lds_off, unsigned long long gaddr) {
  asm volatile("global_load_async_to_lds_b128 %0, %1, off"
               :
               : "v"(lds_off), "v"(gaddr)
               : "memory");
}

// ---------- Kernel 1: merge TT cores into partial tensors A2 / Bm ----------
// A2[m01][r2][n01] = sum_r1 core0[m0*8+r1, n0] * core1[m1*8+r2, r1*8+n1]
// Bm[m23][r2][n23] = sum_r3 core2[m2*8+r3, r2*8+n2] * core3[m3, r3*4+n3]
__global__ void tt_merge_kernel(const float* __restrict__ core0,
                                const float* __restrict__ core1,
                                const float* __restrict__ core2,
                                const float* __restrict__ core3,
                                float* __restrict__ A2,
                                float* __restrict__ Bm) {
  for (int idx = threadIdx.x; idx < 8192; idx += 256) {
    int hiI = idx >> 8;           // m01 / m23  (0..31)
    int r2  = (idx >> 5) & 7;
    int loI = idx & 31;           // n01 / n23  (0..31)

    int m0 = hiI >> 3, m1 = hiI & 7, n0 = loI >> 3, n1 = loI & 7;
    float s = 0.f;
#pragma unroll
    for (int r1 = 0; r1 < 8; ++r1)
      s += core0[(m0 * 8 + r1) * 4 + n0] * core1[(m1 * 8 + r2) * 64 + (r1 * 8 + n1)];
    A2[idx] = s;

    int m2 = hiI >> 2, m3 = hiI & 3, n2 = loI >> 2, n3 = loI & 3;
    float t = 0.f;
#pragma unroll
    for (int r3 = 0; r3 < 8; ++r3)
      t += core2[(m2 * 8 + r3) * 64 + (r2 * 8 + n2)] * core3[m3 * 32 + (r3 * 4 + n3)];
    Bm[idx] = t;
  }
}

// ---------- Kernel 2: W[m,n] = sum_r2 A2[m>>5,r2,n>>5]*Bm[m&31,r2,n&31] -> f16, m-major ----------
__global__ void tt_buildW_kernel(const float* __restrict__ A2,
                                 const float* __restrict__ Bm,
                                 _Float16* __restrict__ Wt) {
  int m = blockIdx.x;
  int mh = (m >> 5) * 256;
  int ml = (m & 31) * 256;
  for (int n = threadIdx.x; n < 1024; n += 256) {
    int nh = n >> 5, nl = n & 31;
    float s = 0.f;
#pragma unroll
    for (int r2 = 0; r2 < 8; ++r2)
      s += A2[mh + r2 * 32 + nh] * Bm[ml + r2 * 32 + nl];
    Wt[(size_t)m * 1024 + n] = (_Float16)s;
  }
}

// ---------- Kernel 3: depthwise 3x3 SAME conv (shared filter), y = conv(x) as f16 ----------
// One block per pixel; all 9-tap branches are block-uniform (scalar branches).
__global__ __launch_bounds__(256)
void tt_conv_kernel(const float* __restrict__ x,     // [16][32][32][1024] f32
                    const float* __restrict__ filt,  // 9 f32
                    _Float16* __restrict__ y) {      // [16384][1024] f16
  const int p = blockIdx.x;
  const int b = p >> 10, h = (p >> 5) & 31, w = p & 31;
  const float* xb = x + ((size_t)b << 20);
  float f[9];
#pragma unroll
  for (int i = 0; i < 9; ++i) f[i] = filt[i];
#pragma unroll
  for (int q = 0; q < 4; ++q) {
    int n = threadIdx.x + q * 256;
    float acc = 0.f;
#pragma unroll
    for (int dh = -1; dh <= 1; ++dh) {
      int hh = h + dh;
      if (hh < 0 || hh > 31) continue;
#pragma unroll
      for (int dw = -1; dw <= 1; ++dw) {
        int ww = w + dw;
        if (ww < 0 || ww > 31) continue;
        acc += f[(dh + 1) * 3 + (dw + 1)] * xb[((hh * 32 + ww) << 10) + n];
      }
    }
    y[((size_t)p << 10) + n] = (_Float16)acc;
  }
}

// ---------- Kernel 4: WMMA GEMM  out[p,m] = relu(y[p,:] . W[m,:] + bias[m]) ----------
// grid.x = 128 pixel-blocks (128 pixels each), grid.y = 2 m-halves (512 m each).
__global__ __launch_bounds__(256)
void tt_gemm_kernel(const _Float16* __restrict__ yglob,  // [16384][1024] f16
                    const _Float16* __restrict__ Wt,     // [1024 m][1024 n] f16
                    const float* __restrict__ bias,      // [1024]
                    float* __restrict__ out) {           // [16384][1024] f32
  extern __shared__ _Float16 yl[];  // [128][LDSTR] f16 (264 KB of the 320 KB WGP LDS)
  const int tid   = threadIdx.x;
  const int pbase = blockIdx.x * 128;

  // ---- phase 1: async-copy the 128x1024 f16 y-tile into padded LDS ----
  // 128 rows x 128 chunks of 16B; LDS row stride 2064B keeps 16B alignment.
  {
    const _Float16* ysrc = yglob + ((size_t)pbase << 10);
    for (int c = tid; c < 16384; c += 256) {
      int row = c >> 7;
      int col = c & 127;
      unsigned lds_off = (unsigned)(size_t)(yl + row * LDSTR + col * 8);
      unsigned long long ga =
          (unsigned long long)(size_t)(ysrc + ((size_t)row << 10) + col * 8);
      async_cp16(lds_off, ga);
    }
    asm volatile("s_wait_asynccnt 0x0" ::: "memory");
    __syncthreads();
  }

  // ---- phase 2: WMMA GEMM ----
  const int wave = tid >> 5;
  const int lane = tid & 31;
  const int wp   = wave >> 2;   // 0..1 : 64-pixel group
  const int wm   = wave & 3;    // 0..3 : 32-m group
  const int ln   = lane & 15;
  const int hi   = lane >> 4;   // half-wave selector

  for (int mc = 0; mc < 4; ++mc) {
    const int mbase = blockIdx.y * 512 + mc * 128 + wm * 32;
    v8f acc[4][2] = {};

#pragma unroll 2
    for (int k = 0; k < 1024; k += 32) {
      // B fragments (32x16, K x N): lanes 0-15 hold K=k..k+15 for col N=ln,
      // lanes 16-31 hold K=k+16..k+31 -> one contiguous 32B load from m-major W.
      v16h bf[2];
#pragma unroll
      for (int j = 0; j < 2; ++j) {
        int col = mbase + j * 16 + ln;
        bf[j] = *(const v16h*)(Wt + (size_t)col * 1024 + k + hi * 16);
      }
      // A fragments (16x32, M x K): lane row = ln; halves 0..15 = K kb..kb+15,
      // kb = k + hi*16  (contiguous, matches B fragment order).
      v16h af[4];
#pragma unroll
      for (int i = 0; i < 4; ++i) {
        const _Float16* rp = yl + (wp * 64 + i * 16 + ln) * LDSTR + k + hi * 16;
        v8h lo = *(const v8h*)(rp);
        v8h up = *(const v8h*)(rp + 8);
        v16h a;
#pragma unroll
        for (int e = 0; e < 8; ++e) { a[e] = lo[e]; a[e + 8] = up[e]; }
        af[i] = a;
      }
#pragma unroll
      for (int i = 0; i < 4; ++i)
#pragma unroll
        for (int j = 0; j < 2; ++j)
          acc[i][j] = __builtin_amdgcn_wmma_f32_16x16x32_f16(
              false, af[i], false, bf[j], (short)0, acc[i][j], false, false);
    }

    // epilogue: C/D layout -> lane col N=ln, rows vg + 8*hi; add bias, ReLU
#pragma unroll
    for (int j = 0; j < 2; ++j) {
      int col = mbase + j * 16 + ln;
      float bc = bias[col];
#pragma unroll
      for (int i = 0; i < 4; ++i) {
        int prow0 = pbase + wp * 64 + i * 16 + hi * 8;
#pragma unroll
        for (int vg = 0; vg < 8; ++vg) {
          float v = acc[i][j][vg] + bc;
          v = v > 0.f ? v : 0.f;
          out[(size_t)(prow0 + vg) * 1024 + col] = v;
        }
      }
    }
  }
}

extern "C" void kernel_launch(void* const* d_in, const int* in_sizes, int n_in,
                              void* d_out, int out_size, void* d_ws, size_t ws_size,
                              hipStream_t stream) {
  (void)in_sizes; (void)n_in; (void)out_size; (void)ws_size;
  const float* x     = (const float*)d_in[0];
  const float* filt  = (const float*)d_in[1];
  const float* core0 = (const float*)d_in[2];
  const float* core1 = (const float*)d_in[3];
  const float* core2 = (const float*)d_in[4];
  const float* core3 = (const float*)d_in[5];
  const float* bias  = (const float*)d_in[6];
  float* out = (float*)d_out;

  // workspace: [0,2MB) Wt f16 ; [2MB,+32KB) A2 ; [+32KB,+64KB) Bm ; [4MB, +33.5MB) y f16
  _Float16* Wt = (_Float16*)d_ws;
  float* A2 = (float*)((char*)d_ws + (2u << 20));
  float* Bm = (float*)((char*)d_ws + (2u << 20) + (32u << 10));
  _Float16* Y = (_Float16*)((char*)d_ws + (4u << 20));

  tt_merge_kernel<<<1, 256, 0, stream>>>(core0, core1, core2, core3, A2, Bm);
  tt_buildW_kernel<<<1024, 256, 0, stream>>>(A2, Bm, Wt);
  tt_conv_kernel<<<16384, 256, 0, stream>>>(x, filt, Y);

  const size_t lds_bytes = (size_t)128 * LDSTR * sizeof(_Float16);  // 264192 B
  hipFuncSetAttribute((const void*)tt_gemm_kernel,
                      hipFuncAttributeMaxDynamicSharedMemorySize, (int)lds_bytes);
  dim3 grid(128, 2);
  tt_gemm_kernel<<<grid, 256, lds_bytes, stream>>>(Y, Wt, bias, out);
}
